// LSTM_82506321756909
// MI455X (gfx1250) — matrix operand order repaired and checked by reading, not compiled, not run on hardware
//
#include <hip/hip_runtime.h>
#include <hip/hip_bf16.h>
#include <stdint.h>

// LSTM scan, B=32, T=2048, D=H=512.  Strategy (MI455X, gfx1250):
//  - W (8MB f32) -> 4MB bf16, pre-swizzled to WMMA B-fragment layout; stays L2-resident.
//  - Persistent 16 WGs x 512 thr (16 wave32 waves).  Per step: [32 x 1024] @ [1024 x 2048]
//    via v_wmma_f32_16x16x32_bf16 (1 16x16 tile per wave, 32 k-slices), f32 accum + bias.
//  - x_{t+1} double-buffered into LDS with GLOBAL_LOAD_ASYNC_TO_LDS_B128 (ASYNCcnt),
//    overlapping the step-t GEMM + cell update + device barrier; only s_wait_asynccnt
//    + an LDS->LDS bf16 convert remain on the critical path.
//  - h exchanged through a 32KB global bf16 buffer + atomic device-wide barrier per step.
//  - x (134MB) and out (134MB) each stream HBM exactly once (~11.5us at 23.3 TB/s).

typedef __attribute__((ext_vector_type(16))) __bf16 v16bf;
typedef __attribute__((ext_vector_type(8)))  __bf16 v8bf;
typedef __attribute__((ext_vector_type(8)))  float  v8f;

#define B_      32
#define T_      2048
#define D_      512
#define H_      512
#define KTOT    1024          // D + H
#define NG      2048          // 4*H
#define NWG     16
#define THREADS 512

// workspace: Wsw (128 gtiles * 32 ks * 32 lanes * 16 bf16 = 4 MiB) | h_g (32 KiB bf16) | ctr
#define WSW_ELEMS (128*32*32*16)

__device__ __forceinline__ __bf16 f2bf(float f) {
  union { float f; uint32_t u; } a; a.f = f;
  uint32_t r = (a.u + 0x7FFFu + ((a.u >> 16) & 1u)) >> 16;   // round-to-nearest-even
  union { unsigned short s; __bf16 b; } o; o.s = (unsigned short)r;
  return o.b;
}

__device__ __forceinline__ float sigf(float v) { return 1.0f / (1.0f + __expf(-v)); }

// Swizzle W[k][col] (f32, row-major K=1024 x N=2048) into per-wave B fragments.
// B 32x16 bf16 layout (wave32): lane%16 -> N col; lanes 0-15 hold K {0..7,16..23},
// lanes 16-31 hold K {8..15,24..31}; element e maps to klocal = (e&7) + 8h + (e>=8 ? 16 : 0).
__global__ void lstm_prep(const float* __restrict__ W, __bf16* __restrict__ Wsw,
                          unsigned int* __restrict__ ctr) {
  long idx = (long)blockIdx.x * blockDim.x + threadIdx.x;
  if (idx == 0) *ctr = 0u;
  if (idx >= (long)WSW_ELEMS) return;
  int e    = (int)(idx & 15);
  int lane = (int)((idx >> 4) & 31);
  int ks   = (int)((idx >> 9) & 31);
  int gt   = (int)(idx >> 14);                 // 0..127 (16-column tile of the 2048 gate cols)
  int col  = gt * 16 + (lane & 15);
  int k    = ks * 32 + (e & 7) + ((lane & 16) ? 8 : 0) + ((e & 8) ? 16 : 0);
  Wsw[idx] = f2bf(W[(long)k * NG + col]);
}

// Issue async global->LDS copies of x[:, t, :] (32 rows x 512 f32 = 64 KiB).
// 4096 B128 ops across 512 threads = exactly 8 per thread (ASYNCcnt += 8).
__device__ __forceinline__ void issue_x_async(const float* __restrict__ x, int t,
                                              float* dstbase /*shared*/, int tid) {
  for (int i = tid; i < (B_ * D_) / 4; i += THREADS) {
    int b  = i >> 7;                                  // 128 float4 per row
    int dv = i & 127;
    const float* g = x + ((long)b * T_ + t) * D_ + dv * 4;
    uint32_t lds = (uint32_t)(uintptr_t)(dstbase + i * 4);  // flat->LDS: addr[31:0]
    asm volatile("global_load_async_to_lds_b128 %0, %1, off"
                 :: "v"(lds), "v"(g) : "memory");
  }
}

__global__ void __launch_bounds__(THREADS)
lstm_scan(const float* __restrict__ x, const float* __restrict__ bias,
          const float* __restrict__ h_init, const __bf16* __restrict__ Wsw,
          __bf16* __restrict__ h_g, unsigned int* __restrict__ ctr,
          float* __restrict__ out) {
  __shared__ __bf16 a_buf[B_ * KTOT];          // 64 KiB: [32 batch][1024 = x|h] bf16
  __shared__ float  xstage[2][B_ * D_];        // 2 x 64 KiB async x double-buffer
  float* gbuf = (float*)a_buf;                 // 16 KiB overlay after GEMM: [32][128] gates

  const int tid   = threadIdx.x;
  const int wg    = blockIdx.x;                // hidden slice [32*wg, 32*wg+32)
  const int wave  = tid >> 5;                  // 0..15
  const int lane  = tid & 31;
  const int mt    = wave >> 3;                 // batch-row tile (0..1)
  const int nt    = wave & 7;                  // local N tile (0..7): [f f i i o o g g]
  const int gate  = nt >> 1;
  const int gtile = 32 * gate + 2 * wg + (nt & 1);
  const int col   = gtile * 16 + (lane & 15);  // this lane's global gate column
  const float bv  = bias[col];

  // Cell state in registers: 2 (batch, hidden) pairs per thread.
  const int p0 = 2 * tid,     b0 = p0 >> 5, j0 = p0 & 31;
  const int p1 = 2 * tid + 1, b1 = p1 >> 5, j1 = p1 & 31;
  float c0 = 0.0f, c1 = 0.0f;

  const v16bf* wp = (const v16bf*)Wsw;
  // 16-bit A 16x32 layout: lanes 0-15 row M=l, K {0..7,16..23}; lanes 16-31 row M=l-16, +8.
  const int a_off0 = (mt * 16 + (lane & 15)) * KTOT + ((lane & 16) ? 8 : 0);

  // Prime the pipeline: x_0 into buffer 0.
  issue_x_async(x, 0, xstage[0], tid);

  for (int t = 0; t < T_; ++t) {
    // ---- kick off x_{t+1} before consuming x_t (stays in flight across the step) ----
    if (t + 1 < T_) {
      issue_x_async(x, t + 1, xstage[(t + 1) & 1], tid);
      // async loads complete in order: <=8 outstanding -> the 8 older (x_t) ops are done
      asm volatile("s_wait_asynccnt 0x8" ::: "memory");
    } else {
      asm volatile("s_wait_asynccnt 0x0" ::: "memory");
    }

    // ---- stage [x_t | h] into a_buf as bf16 (x from LDS staging, h from L2) ----
    {
      const float* xs = xstage[t & 1];
      for (int i = tid; i < B_ * D_; i += THREADS) {
        int b = i >> 9, d = i & 511;
        a_buf[b * KTOT + d] = f2bf(xs[i]);
      }
    }
    if (t == 0) {
      for (int i = tid; i < B_ * H_; i += THREADS) {
        int b = i >> 9, k = i & 511;
        a_buf[b * KTOT + D_ + k] = f2bf(h_init[k]);
      }
    } else {
      for (int i = tid; i < B_ * H_; i += THREADS) {
        int b = i >> 9, k = i & 511;
        a_buf[b * KTOT + D_ + k] = h_g[i];
      }
    }
    __syncthreads();

    // ---- GEMM: one 16x16 tile per wave, K=1024 = 32 x wmma_f32_16x16x32_bf16 ----
    v8f acc = { bv, bv, bv, bv, bv, bv, bv, bv };
#pragma unroll 4
    for (int ks = 0; ks < 32; ++ks) {
      const __bf16* ap = a_buf + a_off0 + ks * 32;
      v8bf alo = *(const v8bf*)(ap);            // K base+{0..7}   (lanes>=16: +8)
      v8bf ahi = *(const v8bf*)(ap + 16);       // K base+{16..23} (lanes>=16: +8)
      v16bf afrag = __builtin_shufflevector(alo, ahi,
          0, 1, 2, 3, 4, 5, 6, 7, 8, 9, 10, 11, 12, 13, 14, 15);
      long widx = ((long)gtile * 32 + ks) * 32 + lane;
      v16bf bfrag = wp[widx];
      if (ks + 1 < 32)
        __builtin_prefetch((const void*)(wp + widx + 32), 0, 1);  // next k-slice (L2)
      acc = __builtin_amdgcn_wmma_f32_16x16x32_bf16(false, afrag, false, bfrag,
                                                    (short)0, acc, false, false);
    }
    __syncthreads();   // all a_buf reads done; safe to overlay gate buffer

    // ---- spill gate tiles to LDS: gbuf[batch][128 = f|i|o|g x 32] ----
    {
      int lc   = nt * 16 + (lane & 15);
      int brow = mt * 16 + ((lane & 16) ? 8 : 0);   // C/D layout: VGPR r -> M = r (+8)
#pragma unroll
      for (int r = 0; r < 8; ++r)
        gbuf[(brow + r) * 128 + lc] = acc[r];
    }
    __syncthreads();

    // ---- LSTM cell update (2 pairs per thread), write h to out + shared h buffer ----
    {
      float f0 = gbuf[b0 * 128 +      j0], i0 = gbuf[b0 * 128 + 32 + j0];
      float o0 = gbuf[b0 * 128 + 64 + j0], g0 = gbuf[b0 * 128 + 96 + j0];
      float f1 = gbuf[b1 * 128 +      j1], i1 = gbuf[b1 * 128 + 32 + j1];
      float o1 = gbuf[b1 * 128 + 64 + j1], g1 = gbuf[b1 * 128 + 96 + j1];
      c0 = sigf(f0) * c0 + sigf(i0) * tanhf(g0);
      c1 = sigf(f1) * c1 + sigf(i1) * tanhf(g1);
      float h0 = sigf(o0) * tanhf(c0);
      float h1 = sigf(o1) * tanhf(c1);
      int col0 = 32 * wg + j0, col1 = 32 * wg + j1;
      out[((long)b0 * T_ + t) * H_ + col0] = h0;
      out[((long)b1 * T_ + t) * H_ + col1] = h1;
      h_g[b0 * H_ + col0] = f2bf(h0);
      h_g[b1 * H_ + col1] = f2bf(h1);
    }

    // ---- device-wide barrier across the 16 persistent workgroups ----
    __threadfence();
    __syncthreads();
    if (tid == 0) {
      __hip_atomic_fetch_add(ctr, 1u, __ATOMIC_RELEASE, __HIP_MEMORY_SCOPE_AGENT);
      unsigned int target = (unsigned int)(NWG * (t + 1));
      while (__hip_atomic_load(ctr, __ATOMIC_ACQUIRE, __HIP_MEMORY_SCOPE_AGENT) < target)
        asm volatile("s_sleep 1");
    }
    __syncthreads();
  }
}

extern "C" void kernel_launch(void* const* d_in, const int* in_sizes, int n_in,
                              void* d_out, int out_size, void* d_ws, size_t ws_size,
                              hipStream_t stream) {
  (void)in_sizes; (void)n_in; (void)out_size; (void)ws_size;
  const float* x      = (const float*)d_in[0];
  /* d_in[1] = input_paddings (unused by reference) */
  const float* W      = (const float*)d_in[2];
  const float* bias   = (const float*)d_in[3];
  const float* h_init = (const float*)d_in[4];

  char* ws = (char*)d_ws;
  __bf16*       Wsw = (__bf16*)ws;                                      // 4 MiB
  __bf16*       h_g = (__bf16*)(ws + (size_t)WSW_ELEMS * 2);            // 32 KiB
  unsigned int* ctr = (unsigned int*)(ws + (size_t)WSW_ELEMS * 2 + 32768);

  lstm_prep<<<WSW_ELEMS / 256, 256, 0, stream>>>(W, Wsw, ctr);
  lstm_scan<<<NWG, THREADS, 0, stream>>>(x, bias, h_init, Wsw, h_g, ctr,
                                         (float*)d_out);
}
